// VocoMorph_85968065396933
// MI455X (gfx1250) — matrix operand order repaired and checked by reading, not compiled, not run on hardware
//
#include <hip/hip_runtime.h>
#include <hip/hip_bf16.h>
#include <math.h>

// ---------------- problem constants ----------------
#define Bn      8
#define Cn      2
#define Tn      262144
#define CHUNKn  8192
#define STRIDEn 6144
#define Kn      43          // ceil(T/STRIDE)
#define NFFTn   1024
#define HOPn    256
#define PADn    512
#define TFn     33          // 1 + CHUNK/HOP
#define Fn      513
#define FPADn   544         // 17*32
#define Hn      512
#define NBn     3
#define EMBn    128
#define Mrows   (Bn*Cn*Kn*TFn)   // 22704
#define ROWS_PER_BC (Kn*TFn)      // 1419
#define LOLAn   (NFFTn + (TFn - 1) * HOPn)  // 9216

typedef __bf16 bf16;
typedef __attribute__((ext_vector_type(16))) __bf16 v16bf;
typedef __attribute__((ext_vector_type(8)))  float  v8f;

union FragU { v16bf v; uint4 q[2]; };

#define WMMA_BF16 __builtin_amdgcn_wmma_f32_16x16x32_bf16

// ---------------- CDNA5 async global->LDS staging ----------------------------
// GLOBAL_LOAD_ASYNC_TO_LDS_B128 (VGLOBAL op 98): VDST = per-lane LDS byte
// offset, VADDR = 64-bit global address. Tracked by ASYNCcnt. The LDS offset is
// the low 32 bits of the generic shared-space address.
__device__ __forceinline__ void async_ld_b128(uint32_t lds_off, uint64_t gaddr) {
  asm volatile("global_load_async_to_lds_b128 %0, %1, off"
               :: "v"(lds_off), "v"(gaddr) : "memory");
}
__device__ __forceinline__ void wait_async0() {
  asm volatile("s_wait_asynccnt 0x0" ::: "memory");
}
__device__ __forceinline__ uint32_t lds_off32(const void* p) {
  return (uint32_t)(uintptr_t)p;
}

// ---------------- fragment loaders (LDS -> VGPR, CDNA5 wave32 layouts) -------
// A tile stored row-major [m][32] bf16. 16-bit A 16x32: lane holds row m=L&15,
// half=L>>4; contiguous k = {half*8..+8} in regs 0..3, {16+half*8..+8} in 4..7.
__device__ __forceinline__ v16bf load_fragA(const bf16* lA, int m, int half) {
  const uint4* p = (const uint4*)(lA + m * 32);
  FragU u;
  u.q[0] = p[half];
  u.q[1] = p[2 + half];
  return u.v;
}
// B tile stored N-major [n][32] bf16 (i.e. W^T). Dense bf16 B 32x16: lane holds
// col n=L&15, half=L>>4; k = {half*8..+8} in regs 0..3, {16+half*8..+8} in 4..7.
__device__ __forceinline__ v16bf load_fragB(const bf16* lB, int n, int half) {
  const uint4* p = (const uint4*)(lB + n * 32);
  FragU u;
  u.q[0] = p[half];
  u.q[1] = p[2 + half];
  return u.v;
}

// ---------------- shared 1024-pt FFT (in LDS, 256 threads) -------------------
__device__ __forceinline__ void fft1024(float* re, float* im, int tid, float sgn) {
  for (int i = tid; i < 1024; i += 256) {
    int j = __brev((unsigned)i) >> 22;
    if (j > i) {
      float tr = re[i]; re[i] = re[j]; re[j] = tr;
      float ti = im[i]; im[i] = im[j]; im[j] = ti;
    }
  }
  __syncthreads();
  for (int len = 2; len <= 1024; len <<= 1) {
    int half = len >> 1;
    float ang0 = sgn * 3.14159265358979323846f / (float)half;
    for (int w = tid; w < 512; w += 256) {
      int grp = w / half;
      int pos = w - grp * half;
      int i0 = grp * len + pos;
      int i1 = i0 + half;
      float s, c;
      __sincosf(ang0 * (float)pos, &s, &c);
      float xr = re[i1], xi = im[i1];
      float tr = xr * c - xi * s;
      float ti = xr * s + xi * c;
      re[i1] = re[i0] - tr; im[i1] = im[i0] - ti;
      re[i0] += tr;         im[i0] += ti;
    }
    __syncthreads();
  }
}

// ---------------- kernel: weight transpose + bf16 convert --------------------
__global__ void prep_kernel(const float* __restrict__ w1, const float* __restrict__ w2,
                            bf16* __restrict__ w1t, bf16* __restrict__ w2t) {
  int idx = blockIdx.x * 256 + threadIdx.x;
  const int SZ = NBn * Hn * FPADn;   // 835584
  if (idx < SZ) {
    int b = idx / (Hn * FPADn);
    int rem = idx - b * (Hn * FPADn);
    int n = rem / FPADn;
    int kk = rem - n * FPADn;
    float v = (kk < Fn) ? w1[(size_t)b * Fn * Hn + (size_t)kk * Hn + n] : 0.0f;
    w1t[idx] = (bf16)v;
  } else if (idx < 2 * SZ) {
    int i2 = idx - SZ;
    int b = i2 / (FPADn * Hn);
    int rem = i2 - b * (FPADn * Hn);
    int n = rem / Hn;
    int kk = rem - n * Hn;
    float v = (n < Fn) ? w2[(size_t)b * Hn * Fn + (size_t)kk * Fn + n] : 0.0f;
    w2t[i2] = (bf16)v;
  }
}

// ---------------- kernel: FiLM (emb @ film_w + film_b) -----------------------
__global__ void film_kernel(const int* __restrict__ effect_id,
                            const float* __restrict__ emb,
                            const float* __restrict__ fw,
                            const float* __restrict__ fb,
                            float* __restrict__ scale, float* __restrict__ shift) {
  int idx = blockIdx.x * 256 + threadIdx.x;           // over B * 2*C*F = 8*2052
  const int JW = 2 * Cn * Fn;                          // 2052
  if (idx >= Bn * JW) return;
  int b = idx / JW, j = idx - b * JW;
  const float* e = emb + (size_t)effect_id[b] * EMBn;
  float s = fb[j];
  #pragma unroll 8
  for (int t = 0; t < EMBn; ++t) s += e[t] * fw[(size_t)t * JW + j];
  if (j < Cn * Fn) scale[(size_t)b * Cn * Fn + j] = s;
  else             shift[(size_t)b * Cn * Fn + (j - Cn * Fn)] = s;
}

// ---------------- kernel: STFT (one frame per workgroup) ---------------------
__global__ __launch_bounds__(256)
void stft_kernel(const float* __restrict__ audio,
                 float* __restrict__ cosph, float* __restrict__ sinph,
                 float* __restrict__ magf, bf16* __restrict__ magbf) {
  __shared__ float re[NFFTn], im[NFFTn];
  const int fi = blockIdx.x;                 // 0..Mrows-1 == ((bc*K)+k)*TF + t
  const int t  = fi % TFn;
  const int kc = (fi / TFn) % Kn;
  const int bc = fi / (TFn * Kn);
  const int tid = threadIdx.x;

  for (int j = tid; j < NFFTn; j += 256) {
    int p = t * HOPn + j;                    // position in reflect-padded chunk
    int q = p - PADn;
    if (q < 0) q = -q;                       // reflect (no edge dup)
    if (q >= CHUNKn) q = 2 * (CHUNKn - 1) - q;
    long s = (long)kc * STRIDEn + q;         // position in zero-padded audio
    float x = (s < Tn) ? audio[(size_t)bc * Tn + s] : 0.0f;
    float w = 0.5f - 0.5f * __cosf(6.283185307179586f * (float)j * (1.0f / NFFTn));
    re[j] = x * w;
    im[j] = 0.0f;
  }
  __syncthreads();
  fft1024(re, im, tid, -1.0f);

  for (int f = tid; f < FPADn; f += 256) {
    float m = 0.0f;
    if (f <= Fn - 1) {
      float xr = re[f], xi = im[f];
      m = sqrtf(xr * xr + xi * xi);
      float cp = 1.0f, sp = 0.0f;
      if (m > 1e-20f) { cp = xr / m; sp = xi / m; }
      size_t o = (size_t)fi * Fn + f;
      cosph[o] = cp; sinph[o] = sp; magf[o] = m;
    }
    magbf[(size_t)fi * FPADn + f] = (bf16)m;   // zero padding kept clean each call
  }
}

// ---------------- kernel: GEMM1  relu(mag @ w1 + b1) -> h (bf16) -------------
// block tile 64M x 256N, 8 waves as 4(M) x 2(N), wave = 16M x 128N (8 wmma tiles)
__global__ __launch_bounds__(256)
void gemm1_kernel(const bf16* __restrict__ magbf, const bf16* __restrict__ w1t,
                  const float* __restrict__ b1, bf16* __restrict__ hbf, int bb) {
  __shared__ bf16 lA[64 * 32];
  __shared__ bf16 lB[256 * 32];
  const int tid  = threadIdx.x;
  const int lane = tid & 31;
  const int wave = tid >> 5;
  const int row0 = blockIdx.x * 64;
  const int col0 = blockIdx.y * 256;
  const int wm = wave & 3;
  const int wn = wave >> 2;
  const int l15 = lane & 15, half = lane >> 4;
  const bf16* w1b = w1t + (size_t)bb * Hn * FPADn;

  v8f acc[8];
  #pragma unroll
  for (int j = 0; j < 8; ++j)
    #pragma unroll
    for (int e = 0; e < 8; ++e) acc[j][e] = 0.0f;

  const int rA = tid >> 2, sA = tid & 3;          // A staging: 1x16B per thread
  const uint32_t ldsA = lds_off32(&((uint4*)lA)[rA * 4 + sA]);
  const bool aValid = (row0 + rA) < Mrows;

  for (int ks = 0; ks < FPADn / 32; ++ks) {
    const int k0 = ks * 32;
    // stage A: 64 rows x 32 k (async global->LDS; zero-fill OOB rows)
    if (aValid) {
      async_ld_b128(ldsA, (uint64_t)(uintptr_t)(magbf + (size_t)(row0 + rA) * FPADn + k0 + sA * 8));
    } else {
      ((uint4*)lA)[rA * 4 + sA] = make_uint4(0u, 0u, 0u, 0u);
    }
    // stage B: 256 n x 32 k (4x16B per thread, async)
    #pragma unroll
    for (int u = 0; u < 4; ++u) {
      int linear = tid + 256 * u;
      int n = linear >> 2, seg = linear & 3;
      async_ld_b128(lds_off32(&((uint4*)lB)[n * 4 + seg]),
                    (uint64_t)(uintptr_t)(w1b + (size_t)(col0 + n) * FPADn + k0 + seg * 8));
    }
    wait_async0();
    __syncthreads();
    v16bf a = load_fragA(lA, wm * 16 + l15, half);
    #pragma unroll
    for (int j = 0; j < 8; ++j) {
      v16bf bf = load_fragB(lB, wn * 128 + j * 16 + l15, half);
      acc[j] = WMMA_BF16(false, a, false, bf, (short)0, acc[j], false, false);
    }
    __syncthreads();
  }
  // epilogue: bias + relu -> bf16
  #pragma unroll
  for (int j = 0; j < 8; ++j) {
    int n = col0 + wn * 128 + j * 16 + l15;
    float bias = b1[(size_t)bb * Hn + n];
    #pragma unroll
    for (int r = 0; r < 8; ++r) {
      int grow = row0 + wm * 16 + half * 8 + r;
      if (grow < Mrows) {
        float v = acc[j][r] + bias;
        v = v > 0.0f ? v : 0.0f;
        hbf[(size_t)grow * Hn + n] = (bf16)v;
      }
    }
  }
}

// ---------------- kernel: GEMM2  relu((h@w2+b2)*scale+shift+res) -> mag ------
// block tile 64M x 128N, 8 waves as 4(M) x 2(N), wave = 16M x 64N (4 wmma tiles)
__global__ __launch_bounds__(256)
void gemm2_kernel(const bf16* __restrict__ hbf, const bf16* __restrict__ w2t,
                  const float* __restrict__ b2,
                  const float* __restrict__ scale, const float* __restrict__ shift,
                  float* __restrict__ magf, bf16* __restrict__ magbf, int bb) {
  __shared__ bf16 lA[64 * 32];
  __shared__ bf16 lB[128 * 32];
  const int tid  = threadIdx.x;
  const int lane = tid & 31;
  const int wave = tid >> 5;
  const int row0 = blockIdx.x * 64;
  const int col0 = blockIdx.y * 128;
  const int wm = wave & 3;
  const int wn = wave >> 2;
  const int l15 = lane & 15, half = lane >> 4;
  const bf16* w2b = w2t + (size_t)bb * FPADn * Hn;

  v8f acc[4];
  #pragma unroll
  for (int j = 0; j < 4; ++j)
    #pragma unroll
    for (int e = 0; e < 8; ++e) acc[j][e] = 0.0f;

  const int rA = tid >> 2, sA = tid & 3;
  const uint32_t ldsA = lds_off32(&((uint4*)lA)[rA * 4 + sA]);
  const bool aValid = (row0 + rA) < Mrows;

  for (int ks = 0; ks < Hn / 32; ++ks) {
    const int k0 = ks * 32;
    // stage A (async; zero-fill OOB rows)
    if (aValid) {
      async_ld_b128(ldsA, (uint64_t)(uintptr_t)(hbf + (size_t)(row0 + rA) * Hn + k0 + sA * 8));
    } else {
      ((uint4*)lA)[rA * 4 + sA] = make_uint4(0u, 0u, 0u, 0u);
    }
    // stage B: 128 n x 32 k (2x16B per thread; zero-fill cols beyond FPADn)
    #pragma unroll
    for (int u = 0; u < 2; ++u) {
      int linear = tid + 256 * u;
      int n = linear >> 2, seg = linear & 3;
      int gn = col0 + n;
      if (gn < FPADn) {
        async_ld_b128(lds_off32(&((uint4*)lB)[n * 4 + seg]),
                      (uint64_t)(uintptr_t)(w2b + (size_t)gn * Hn + k0 + seg * 8));
      } else {
        ((uint4*)lB)[n * 4 + seg] = make_uint4(0u, 0u, 0u, 0u);
      }
    }
    wait_async0();
    __syncthreads();
    v16bf a = load_fragA(lA, wm * 16 + l15, half);
    #pragma unroll
    for (int j = 0; j < 4; ++j) {
      v16bf bf = load_fragB(lB, wn * 64 + j * 16 + l15, half);
      acc[j] = WMMA_BF16(false, a, false, bf, (short)0, acc[j], false, false);
    }
    __syncthreads();
  }
  // epilogue: FiLM + residual + relu
  #pragma unroll
  for (int j = 0; j < 4; ++j) {
    int n = col0 + wn * 64 + j * 16 + l15;
    if (n < Fn) {
      float bias = b2[(size_t)bb * Fn + n];
      #pragma unroll
      for (int r = 0; r < 8; ++r) {
        int grow = row0 + wm * 16 + half * 8 + r;
        if (grow < Mrows) {
          int bc = grow / ROWS_PER_BC;
          float sc = scale[(size_t)bc * Fn + n];
          float sh = shift[(size_t)bc * Fn + n];
          float res = magf[(size_t)grow * Fn + n];
          float v = (acc[j][r] + bias) * sc + sh + res;
          v = v > 0.0f ? v : 0.0f;
          magf[(size_t)grow * Fn + n]  = v;
          magbf[(size_t)grow * FPADn + n] = (bf16)v;
        }
      }
    }
  }
}

// ---------------- kernel: iSTFT + intra-chunk OLA + env normalize ------------
__global__ __launch_bounds__(256)
void istft_kernel(const float* __restrict__ magf, const float* __restrict__ cosph,
                  const float* __restrict__ sinph, float* __restrict__ chunk_audio) {
  __shared__ float ola[LOLAn];
  __shared__ float re[NFFTn], im[NFFTn];
  const int cb  = blockIdx.x;        // bc*K + k, 0..687
  const int tid = threadIdx.x;

  for (int j = tid; j < LOLAn; j += 256) ola[j] = 0.0f;
  __syncthreads();

  for (int t = 0; t < TFn; ++t) {
    size_t row = (size_t)cb * TFn + t;
    for (int f = tid; f <= Fn - 1; f += 256) {
      size_t o = row * Fn + f;
      float m  = magf[o];
      float xr = m * cosph[o];
      float xi = m * sinph[o];
      re[f] = xr; im[f] = xi;
      if (f >= 1 && f <= NFFTn / 2 - 1) { re[NFFTn - f] = xr; im[NFFTn - f] = -xi; }
    }
    __syncthreads();
    fft1024(re, im, tid, +1.0f);
    for (int j = tid; j < NFFTn; j += 256) {
      float w = 0.5f - 0.5f * __cosf(6.283185307179586f * (float)j * (1.0f / NFFTn));
      ola[t * HOPn + j] += re[j] * (1.0f / NFFTn) * w;
    }
    __syncthreads();
  }

  for (int j = tid; j < CHUNKn; j += 256) {
    int p = j + PADn;
    float env = 0.0f;
    int tmin = (p > NFFTn - 1) ? (p - (NFFTn - 1) + HOPn - 1) / HOPn : 0;
    int tmax = p / HOPn; if (tmax > TFn - 1) tmax = TFn - 1;
    for (int tt = tmin; tt <= tmax; ++tt) {
      int x = p - tt * HOPn;
      float w = 0.5f - 0.5f * __cosf(6.283185307179586f * (float)x * (1.0f / NFFTn));
      env += w * w;
    }
    float s = ola[p] / (env > 1e-8f ? env : 1.0f);
    chunk_audio[(size_t)cb * CHUNKn + j] = s;
  }
}

// ---------------- kernel: inter-chunk overlap-add (gather, no atomics) -------
__global__ void ola_kernel(const float* __restrict__ chunk_audio, float* __restrict__ out) {
  long idx = (long)blockIdx.x * 256 + threadIdx.x;    // over B*C*T
  if (idx >= (long)Bn * Cn * Tn) return;
  int bc   = (int)(idx / Tn);
  int tpos = (int)(idx - (long)bc * Tn);
  int kmax = tpos / STRIDEn; if (kmax > Kn - 1) kmax = Kn - 1;
  int kmin = (tpos >= CHUNKn) ? (tpos - CHUNKn) / STRIDEn + 1 : 0;
  float s = 0.0f;
  for (int k = kmin; k <= kmax; ++k)
    s += chunk_audio[((size_t)bc * Kn + k) * CHUNKn + (tpos - k * STRIDEn)];
  out[idx] = s;
}

// ---------------- launch -----------------------------------------------------
extern "C" void kernel_launch(void* const* d_in, const int* in_sizes, int n_in,
                              void* d_out, int out_size, void* d_ws, size_t ws_size,
                              hipStream_t stream) {
  const float* audio   = (const float*)d_in[0];
  const int*   eid     = (const int*)  d_in[1];
  const float* embt    = (const float*)d_in[2];
  const float* film_w  = (const float*)d_in[3];
  const float* film_b  = (const float*)d_in[4];
  const float* rb_w1   = (const float*)d_in[5];
  const float* rb_b1   = (const float*)d_in[6];
  const float* rb_w2   = (const float*)d_in[7];
  const float* rb_b2   = (const float*)d_in[8];
  float* out = (float*)d_out;

  size_t off = 0;
  auto carve = [&](size_t bytes) -> void* {
    void* p = (char*)d_ws + off;
    off += (bytes + 255) & ~(size_t)255;
    return p;
  };
  float* cosph = (float*)carve((size_t)Mrows * Fn * sizeof(float));
  float* sinph = (float*)carve((size_t)Mrows * Fn * sizeof(float));
  float* magf  = (float*)carve((size_t)Mrows * Fn * sizeof(float));
  bf16*  magbf = (bf16*) carve((size_t)Mrows * FPADn * sizeof(bf16));
  bf16*  hbf   = (bf16*) carve((size_t)Mrows * Hn * sizeof(bf16));
  bf16*  w1t   = (bf16*) carve((size_t)NBn * Hn * FPADn * sizeof(bf16));
  bf16*  w2t   = (bf16*) carve((size_t)NBn * FPADn * Hn * sizeof(bf16));
  float* scale = (float*)carve((size_t)Bn * Cn * Fn * sizeof(float));
  float* shift = (float*)carve((size_t)Bn * Cn * Fn * sizeof(float));
  float* chka  = (float*)carve((size_t)Bn * Cn * Kn * CHUNKn * sizeof(float));
  (void)ws_size; (void)n_in; (void)in_sizes; (void)out_size;

  // 1. weight prep (transpose + bf16, zero-padded)
  {
    int total = 2 * NBn * Hn * FPADn;
    prep_kernel<<<(total + 255) / 256, 256, 0, stream>>>(rb_w1, rb_w2, w1t, w2t);
  }
  // 2. FiLM
  {
    int total = Bn * 2 * Cn * Fn;
    film_kernel<<<(total + 255) / 256, 256, 0, stream>>>(eid, embt, film_w, film_b, scale, shift);
  }
  // 3. STFT
  stft_kernel<<<Mrows, 256, 0, stream>>>(audio, cosph, sinph, magf, magbf);

  // 4. residual blocks (WMMA GEMMs)
  dim3 g1((Mrows + 63) / 64, Hn / 256);   // 355 x 2
  dim3 g2((Mrows + 63) / 64, 5);          // 355 x 5 (covers 513 cols with guards)
  for (int bb = 0; bb < NBn; ++bb) {
    gemm1_kernel<<<g1, 256, 0, stream>>>(magbf, w1t, rb_b1, hbf, bb);
    gemm2_kernel<<<g2, 256, 0, stream>>>(hbf, w2t, rb_b2, scale, shift, magf, magbf, bb);
  }

  // 5. iSTFT + intra-chunk OLA
  istft_kernel<<<Bn * Cn * Kn, 256, 0, stream>>>(magf, cosph, sinph, chka);

  // 6. inter-chunk overlap-add (writes every output sample)
  {
    long total = (long)Bn * Cn * Tn;
    ola_kernel<<<(int)((total + 255) / 256), 256, 0, stream>>>(chka, out);
  }
}
